// RTMDetPost_25769803776044
// MI455X (gfx1250) — matrix operand order repaired and checked by reading, not compiled, not run on hardware
//
#include <hip/hip_runtime.h>
#include <hip/hip_bf16.h>

typedef __attribute__((ext_vector_type(16))) _Float16 v16h;
typedef __attribute__((ext_vector_type(8)))  _Float16 v8h;
typedef __attribute__((ext_vector_type(8)))  float    v8f;
typedef __attribute__((ext_vector_type(4)))  int      v4i;

#define P_TOT   8400
#define NCLS    80
#define KDIM    169
#define PRED_W  (NCLS + 4 + KDIM)   // 253
#define KPRE    1000
#define MAXD    100
#define Hm      80
#define Wm      80
#define HWm     6400
#define UPD     640
#define NMS_PRE 0.05f
#define NMS_T   0.6f

// d_out layout (floats): boxes(400) | scores(100) | labels(100) | masks(40,960,000) | valid(100)
#define OUT_BOXES  0
#define OUT_SCORES 400
#define OUT_LABELS 500
#define OUT_MASKS  600
#define OUT_VALID  (600 + MAXD * UPD * UPD)

// ---- CDNA5 async global->LDS path (builtin confirmed present; signature: generic int4* params)
#if defined(__has_builtin)
#  if __has_builtin(__builtin_amdgcn_global_load_async_to_lds_b128)
#    define HAVE_ASYNC_LDS 1
#  endif
#endif
#ifndef HAVE_ASYNC_LDS
#  define HAVE_ASYNC_LDS 0
#endif

__device__ __forceinline__ void async_copy16(const v8h* gsrc, v8h* ldst) {
#if HAVE_ASYNC_LDS
  __builtin_amdgcn_global_load_async_to_lds_b128((v4i*)gsrc, (v4i*)ldst, 0, 0);
#else
  *ldst = *gsrc;   // global_load_b128 + ds_store_b128 fallback
#endif
}

__device__ __forceinline__ void wait_async_done() {
#if HAVE_ASYNC_LDS
#  if __has_builtin(__builtin_amdgcn_s_wait_asynccnt)
  __builtin_amdgcn_s_wait_asynccnt(0);
#  else
  asm volatile("s_wait_asynccnt 0" ::: "memory");
#  endif
#endif
}

// ---------------- 1: per-anchor max class score (sigmoid monotonic -> argmax of logits)
__global__ void k_scores(const float* __restrict__ preds,
                         float* __restrict__ scores, int* __restrict__ labels) {
  int i = blockIdx.x * blockDim.x + threadIdx.x;
  if (i >= P_TOT) return;
  const float* row = preds + (size_t)i * PRED_W;
  float m = row[0]; int l = 0;
  for (int c = 1; c < NCLS; ++c) { float v = row[c]; if (v > m) { m = v; l = c; } }
  scores[i] = 1.0f / (1.0f + __expf(-m));
  labels[i] = l;
}

// ---------------- 2: exact top-K via rank counting (stable, index tiebreak)
__global__ void k_topk(const float* __restrict__ scores,
                       int* __restrict__ topk_idx, float* __restrict__ top_scores) {
  __shared__ float s[P_TOT];  // 33.6 KB
  for (int j = threadIdx.x; j < P_TOT; j += blockDim.x) s[j] = scores[j];
  __syncthreads();
  int i = blockIdx.x * blockDim.x + threadIdx.x;
  if (i >= P_TOT) return;
  float mi = s[i];
  int rank = 0;
  for (int j = 0; j < P_TOT; ++j) {
    float v = s[j];
    rank += (v > mi) || (v == mi && j < i);
  }
  if (rank < KPRE) { topk_idx[rank] = i; top_scores[rank] = mi; }
}

// ---------------- 3: decode boxes + gather labels
__global__ void k_decode(const float* __restrict__ preds, const float* __restrict__ priors,
                         const int* __restrict__ topk_idx, const int* __restrict__ labels_all,
                         float* __restrict__ boxes, int* __restrict__ lbl) {
  int k = blockIdx.x * blockDim.x + threadIdx.x;
  if (k >= KPRE) return;
  int id = topk_idx[k];
  const float* row = preds + (size_t)id * PRED_W + NCLS;
  float px = priors[id * 4 + 0], py = priors[id * 4 + 1];
  boxes[k * 4 + 0] = px - row[0];
  boxes[k * 4 + 1] = py - row[1];
  boxes[k * 4 + 2] = px + row[2];
  boxes[k * 4 + 3] = py + row[3];
  lbl[k] = labels_all[id];
}

// ---------------- 4: greedy class-offset NMS, single block
__global__ void k_nms(const float* __restrict__ boxes, const int* __restrict__ lbl,
                      const float* __restrict__ top_scores, int* __restrict__ keep) {
  __shared__ float bx0[KPRE], by0[KPRE], bx1[KPRE], by1[KPRE], ar[KPRE];
  __shared__ int   kp[KPRE];
  __shared__ float smax[256];
  int t = threadIdx.x;
  float lm = -3.4e38f;
  for (int j = t; j < KPRE; j += 256) {
    float x0 = boxes[j*4+0], y0 = boxes[j*4+1], x1 = boxes[j*4+2], y1 = boxes[j*4+3];
    bx0[j]=x0; by0[j]=y0; bx1[j]=x1; by1[j]=y1;
    lm = fmaxf(lm, fmaxf(fmaxf(x0, y0), fmaxf(x1, y1)));
  }
  smax[t] = lm; __syncthreads();
  for (int s = 128; s > 0; s >>= 1) { if (t < s) smax[t] = fmaxf(smax[t], smax[t+s]); __syncthreads(); }
  float offu = smax[0] + 1.0f;
  for (int j = t; j < KPRE; j += 256) {
    float o = (float)lbl[j] * offu;
    bx0[j]+=o; by0[j]+=o; bx1[j]+=o; by1[j]+=o;
    ar[j] = fmaxf(bx1[j]-bx0[j], 0.0f) * fmaxf(by1[j]-by0[j], 0.0f);
    kp[j] = (top_scores[j] > NMS_PRE) ? 1 : 0;
  }
  __syncthreads();
  for (int i = 0; i < KPRE; ++i) {
    if (kp[i]) {  // uniform across block
      float ix0=bx0[i], iy0=by0[i], ix1=bx1[i], iy1=by1[i], ia=ar[i];
      for (int j = i + 1 + t; j < KPRE; j += 256) {
        if (kp[j]) {
          float ltx = fmaxf(ix0, bx0[j]), lty = fmaxf(iy0, by0[j]);
          float rbx = fminf(ix1, bx1[j]), rby = fminf(iy1, by1[j]);
          float inter = fmaxf(rbx - ltx, 0.0f) * fmaxf(rby - lty, 0.0f);
          float iou = inter / (ia + ar[j] - inter + 1e-7f);
          if (iou > NMS_T) kp[j] = 0;
        }
      }
    }
    __syncthreads();
  }
  for (int j = t; j < KPRE; j += 256) keep[j] = kp[j];
}

// ---------------- 5: stable partition (argsort(~keep))[:100] + gather det outputs
__global__ void k_order(const int* __restrict__ keep, const int* __restrict__ topk_idx,
                        const float* __restrict__ top_scores, const int* __restrict__ lbl,
                        const float* __restrict__ boxes,
                        float* __restrict__ out, int* __restrict__ sel_idx, int* __restrict__ dval) {
  __shared__ int kp[KPRE];
  __shared__ int total;
  int t = threadIdx.x;
  for (int j = t; j < KPRE; j += blockDim.x) kp[j] = keep[j];
  __syncthreads();
  if (t == 0) { int s = 0; for (int j = 0; j < KPRE; ++j) s += kp[j]; total = s; }
  __syncthreads();
  if (t < KPRE) {
    int cb = 0, nb = 0;
    for (int j = 0; j < t; ++j) { if (kp[j]) cb++; else nb++; }
    int pos = kp[t] ? cb : (total + nb);
    if (pos < MAXD) {
      sel_idx[pos] = topk_idx[t];
      dval[pos] = kp[t];
      out[OUT_BOXES + pos*4 + 0] = boxes[t*4+0];
      out[OUT_BOXES + pos*4 + 1] = boxes[t*4+1];
      out[OUT_BOXES + pos*4 + 2] = boxes[t*4+2];
      out[OUT_BOXES + pos*4 + 3] = boxes[t*4+3];
      out[OUT_SCORES + pos] = top_scores[t];
      out[OUT_LABELS + pos] = (float)lbl[t];
      out[OUT_VALID  + pos] = kp[t] ? 1.0f : 0.0f;
    }
  }
}

// ---------------- 6a: pack mask feats (8,80,80) f32 -> pixel-major (6400,8) f16
// One b128 load then yields all 8 channels of one pixel (matches WMMA B-operand lane layout).
__global__ void k_packmf(const float* __restrict__ mf, v8h* __restrict__ mfT) {
  int p = blockIdx.x * blockDim.x + threadIdx.x;
  if (p >= HWm) return;
  v8h v;
  #pragma unroll
  for (int ch = 0; ch < 8; ++ch) v[ch] = (_Float16)mf[ch * HWm + p];
  mfT[p] = v;
}

// ---------------- 6b: mask head, chained WMMA + double-buffered async global->LDS staging
__global__ void k_maskhead(const float* __restrict__ preds, const float* __restrict__ priors,
                           const int* __restrict__ sel_idx, const v8h* __restrict__ mfT,
                           float* __restrict__ logits) {
  constexpr int CHUNK  = 1600;          // pixels per staged chunk (25.6 KB per buffer)
  constexpr int NCHUNK = HWm / CHUNK;   // 4
  __shared__ v8h smf[2][CHUNK];         // 51.2 KB double buffer

  const int d    = blockIdx.x;
  const int tid  = threadIdx.x;
  const int lane = tid & 31;
  const int wid  = tid >> 5;
  const int nwav = blockDim.x >> 5;
  const int id   = sel_idx[d];
  const float* kn = preds + (size_t)id * PRED_W + NCLS + 4;   // 169 kernel params
  const float prx = priors[id*4 + 0];
  const float pry = priors[id*4 + 1];
  const float inv = 1.0f / (priors[id*4 + 2] * 8.0f);

  const int  M  = lane & 15;
  const bool lo = lane < 16;
  const int  kh = lo ? 0 : 8;   // A-matrix K-half per lane group

  // A operands: 16x32 f16 layout (contiguous: lane<16 -> K 0..15 in elements 0..15, lane>=16 -> K 16..31)
  v16h a1, a2;
  #pragma unroll
  for (int j = 0; j < 16; ++j) { a1[j] = (_Float16)0.0f; a2[j] = (_Float16)0.0f; }
  #pragma unroll
  for (int j = 0; j < 10; ++j) {
    int kk = j;
    float w1v = (M < 8 && lo) ? kn[M * 10 + kk] : 0.0f;            // w1: 8x10
    a1[j] = (_Float16)w1v;
    if (j < 8) {
    float w2v = (M < 8 && lo) ? kn[80 + M * 8 + j] : 0.0f;              // w2: 8x8
    a2[j] = (_Float16)w2v;
    }
  }
  float bb1[8], bb2[8], w3[8];
  #pragma unroll
  for (int o = 0; o < 8; ++o) { w3[o] = kn[144+o]; bb1[o] = kn[152+o]; bb2[o] = kn[160+o]; }
  const float b3 = kn[168];

  float* lg = logits + (size_t)d * HWm;
  const v8f cz = {0.f,0.f,0.f,0.f,0.f,0.f,0.f,0.f};

  // prime buffer 0 with chunk 0
  for (int pix = tid; pix < CHUNK; pix += blockDim.x)
    async_copy16(mfT + pix, &smf[0][pix]);
  wait_async_done();
  __syncthreads();

  for (int c = 0; c < NCHUNK; ++c) {
    const int buf   = c & 1;
    const int cbase = c * CHUNK;
    // kick off DMA of next chunk into the other buffer (overlaps with WMMA below)
    if (c + 1 < NCHUNK) {
      for (int pix = tid; pix < CHUNK; pix += blockDim.x)
        async_copy16(mfT + cbase + CHUNK + pix, &smf[buf ^ 1][pix]);
      __builtin_prefetch(mfT + cbase + 2 * CHUNK, 0, 1);  // global_prefetch_b8 toward chunk c+2
    }

    // compute this chunk: 100 tiles of 16 pixels, split across waves
    for (int t = wid; t < CHUNK / 16; t += nwav) {
      int pl = t * 16 + M;        // pixel within chunk (dup in upper half-wave)
      int p  = cbase + pl;        // global pixel
      v8h mv = smf[buf][pl];      // ds_load_b128: all 8 channels of this pixel
      float fx = (float)(p % Wm) * 8.0f;
      float fy = (float)(p / Wm) * 8.0f;
      // B operand: 32x16 f16; lane<16 -> N=lane, halves K=0..15 (channels 0..9 live, rest 0)
      v16h b;
      #pragma unroll
      for (int j = 0; j < 16; ++j) b[j] = (_Float16)0.0f;
      b[0] = lo ? (_Float16)((prx - fx) * inv) : (_Float16)0.0f;
      b[1] = lo ? (_Float16)((pry - fy) * inv) : (_Float16)0.0f;
      #pragma unroll
      for (int ch = 0; ch < 8; ++ch) b[2 + ch] = lo ? mv[ch] : (_Float16)0.0f;
      // layer 1: 8x10 GEMM tile
      v8f c1 = __builtin_amdgcn_wmma_f32_16x16x32_f16(false, a1, false, b, (short)0, cz, false, false);
      // bias+relu; f32 C layout == f16 B layout per-lane -> no cross-lane traffic
      v16h y;
      #pragma unroll
      for (int j = 0; j < 16; ++j) y[j] = (_Float16)0.0f;
      #pragma unroll
      for (int o = 0; o < 8; ++o) {
        float v = fmaxf(c1[o] + bb1[o], 0.0f);
        y[o] = lo ? (_Float16)v : (_Float16)0.0f;
      }
      // layer 2: 8x8 GEMM tile
      v8f c2 = __builtin_amdgcn_wmma_f32_16x16x32_f16(false, a2, false, y, (short)0, cz, false, false);
      // layer 3: 1x8 dot in VALU
      float acc = b3;
      #pragma unroll
      for (int i = 0; i < 8; ++i) acc += w3[i] * fmaxf(c2[i] + bb2[i], 0.0f);
      if (lo) lg[p] = acc;
    }

    if (c + 1 < NCHUNK) wait_async_done();  // my async transfers for chunk c+1 done
    __syncthreads();                        // everyone done reading buf / writing buf^1
  }
}

// ---------------- 7: 8x bilinear upsample + sigmoid>0.5 (== logit>0) + valid mask
__global__ void k_up(const float* __restrict__ logits, const int* __restrict__ dval,
                     float* __restrict__ out) {
  unsigned gid = blockIdx.x * blockDim.x + threadIdx.x;
  const unsigned TOT = (unsigned)MAXD * UPD * UPD;
  if (gid >= TOT) return;
  int d   = gid / (UPD * UPD);
  int rem = gid % (UPD * UPD);
  int r = rem / UPD, c = rem % UPD;
  float ir = fminf(fmaxf((r + 0.5f) * 0.125f - 0.5f, 0.0f), 79.0f);
  float ic = fminf(fmaxf((c + 0.5f) * 0.125f - 0.5f, 0.0f), 79.0f);
  int r0 = (int)ir, c0 = (int)ic;
  int r1 = r0 < 79 ? r0 + 1 : 79;
  int c1 = c0 < 79 ? c0 + 1 : 79;
  float wr = ir - (float)r0, wc = ic - (float)c0;
  const float* L = logits + (size_t)d * HWm;
  float v00 = L[r0 * Wm + c0], v01 = L[r0 * Wm + c1];
  float v10 = L[r1 * Wm + c0], v11 = L[r1 * Wm + c1];
  float v = v00 * (1.0f - wr) * (1.0f - wc) + v01 * (1.0f - wr) * wc
          + v10 * wr * (1.0f - wc)          + v11 * wr * wc;
  out[OUT_MASKS + gid] = (v > 0.0f && dval[d]) ? 1.0f : 0.0f;
}

extern "C" void kernel_launch(void* const* d_in, const int* in_sizes, int n_in,
                              void* d_out, int out_size, void* d_ws, size_t ws_size,
                              hipStream_t stream) {
  const float* preds  = (const float*)d_in[0];  // (1, 8400, 253)
  const float* mf     = (const float*)d_in[1];  // (1, 8, 80, 80)
  const float* priors = (const float*)d_in[2];  // (8400, 4)
  // d_in[3] = coord: recomputed on the fly
  float* out = (float*)d_out;

  // workspace carve-up (float units)
  float* ws          = (float*)d_ws;
  float* scores_all  = ws;                    // 8400
  int*   labels_all  = (int*)(ws + 8400);     // 8400
  int*   topk_idx    = (int*)(ws + 16800);    // 1000
  float* top_scores  = ws + 17800;            // 1000
  float* boxes       = ws + 18800;            // 4000
  int*   lbl         = (int*)(ws + 22800);    // 1000
  int*   keep        = (int*)(ws + 23800);    // 1000
  int*   sel_idx     = (int*)(ws + 24800);    // 100
  int*   dval        = (int*)(ws + 24900);    // 100
  float* logits      = ws + 25008;            // 640000  (100 x 80 x 80)
  v8h*   mfT         = (v8h*)(ws + 665008);   // 6400 x 16B = 102400 B (16B aligned)

  k_scores<<<(P_TOT + 255) / 256, 256, 0, stream>>>(preds, scores_all, labels_all);
  k_topk<<<(P_TOT + 255) / 256, 256, 0, stream>>>(scores_all, topk_idx, top_scores);
  k_decode<<<(KPRE + 255) / 256, 256, 0, stream>>>(preds, priors, topk_idx, labels_all, boxes, lbl);
  k_nms<<<1, 256, 0, stream>>>(boxes, lbl, top_scores, keep);
  k_order<<<1, 1024, 0, stream>>>(keep, topk_idx, top_scores, lbl, boxes, out, sel_idx, dval);
  k_packmf<<<(HWm + 255) / 256, 256, 0, stream>>>(mf, mfT);
  k_maskhead<<<MAXD, 256, 0, stream>>>(preds, priors, sel_idx, mfT, logits);
  const unsigned TOT = (unsigned)MAXD * UPD * UPD;
  k_up<<<(TOT + 255) / 256, 256, 0, stream>>>(logits, dval, out);
}
